// CustomConv2D_57200374448719
// MI455X (gfx1250) — compile-verified, matrix operand order checked by reading, not run on hardware
//
#include <hip/hip_runtime.h>

// CDNA5 (gfx1250) implicit-GEMM conv: bf16 WMMA, double-buffered LDS,
// async global->LDS weight staging, branch-free im2col gather.
// GEMM: M=Cout=256, N=B*62*62, K ordered k = pos*128 + c  (pos = kh*3+kw).

typedef __attribute__((ext_vector_type(16))) __bf16        v16bf;
typedef __attribute__((ext_vector_type(8)))  float         v8f;
typedef __attribute__((ext_vector_type(4)))  unsigned int  u32x4;

#define BM   128
#define BN   128
#define LDT  40          // padded LDS row stride (elems) -> 80 bytes
#define TILEB (BM * LDT * 2)   // bytes per LDS tile buffer (10240)
#define KTOT 1152
#define NSP  3844
#define COUT 256
#define CIN  128
#define IH   64
#define IW   64
#define OWD  62
#define HW   (IH * IW)   // 4096
#define NSTEP 36         // 9 taps * 4 Cin-chunks of 32
#define DATA_ELEMS (32u * 128u * 64u * 64u)
#define WGT_ELEMS  (256u * 128u * 9u)

__device__ __forceinline__ unsigned short f2bf(float f) {
  unsigned int u = __float_as_uint(f);
  u += 0x7FFFu + ((u >> 16) & 1u);
  return (unsigned short)(u >> 16);
}

// ---- pre-pass 1: weights fp32 OIHW -> bf16 [o][pos][c] ----------------------
__global__ __launch_bounds__(256)
void wgt_to_bf16(const float* __restrict__ w, unsigned short* __restrict__ dst) {
  const unsigned i = blockIdx.x * 256u + threadIdx.x;
  const unsigned o = i / KTOT;
  const unsigned r = i - o * KTOT;
  const unsigned pos = r >> 7;
  const unsigned c   = r & 127;
  dst[(size_t)o * KTOT + pos * CIN + c] = f2bf(w[(size_t)o * KTOT + c * 9 + pos]);
}

// ---- pre-pass 2: data fp32 -> bf16, same NCHW layout ------------------------
__global__ __launch_bounds__(256)
void data_to_bf16(const float* __restrict__ src, unsigned short* __restrict__ dst) {
  const size_t i = ((size_t)blockIdx.x * 256u + threadIdx.x) * 4u;
  float4 f = *(const float4*)(src + i);
  unsigned lo = (unsigned)f2bf(f.x) | ((unsigned)f2bf(f.y) << 16);
  unsigned hi = (unsigned)f2bf(f.z) | ((unsigned)f2bf(f.w) << 16);
  *(uint2*)(dst + i) = make_uint2(lo, hi);
}

union FragU { u32x4 q[2]; v16bf v; };

// ---- main kernel ------------------------------------------------------------
__global__ __launch_bounds__(256)
void conv2d_wmma_bf16(const unsigned short* __restrict__ dataBf,
                      const unsigned short* __restrict__ wgtBf,
                      float* __restrict__ out) {
  __shared__ unsigned short sA[2][BM * LDT];
  __shared__ unsigned short sB[2][BN * LDT];

  const int tid  = threadIdx.x;
  const int lane = tid & 31;
  const int wave = tid >> 5;
  const int wm   = wave >> 1;       // 0..3 : 32-row Cout slab
  const int wn   = wave & 1;        // 0..1 : 64-col spatial slab
  const int half = lane >> 4;
  const int lrow = lane & 15;

  const int n0 = blockIdx.x * BN;
  const int m0 = blockIdx.y * BM;
  const int b  = blockIdx.z;

  // staging coords: 2 threads per row/col, 16 K-elems each
  const int lr = tid >> 1;
  const int lh = tid & 1;

  // Clamp out-of-range columns to (0,0): addresses stay in-bounds; the
  // garbage lands only in output columns the epilogue never stores.
  const int nIdx = n0 + lr;
  const int ohc  = (nIdx < NSP) ? (nIdx / OWD) : 0;
  const int owc  = (nIdx < NSP) ? (nIdx - ohc * OWD) : 0;

  const unsigned ldsABase = (unsigned)(uintptr_t)(&sA[0][0] + lr * LDT + lh * 16);

  v8f acc[2][4];
  const v8f vzero = {0.f, 0.f, 0.f, 0.f, 0.f, 0.f, 0.f, 0.f};
#pragma unroll
  for (int i = 0; i < 2; i++)
#pragma unroll
    for (int j = 0; j < 4; j++) acc[i][j] = vzero;

  // stage tile for step s2 into LDS buffer `buf` (issue only; no waits here)
  auto stage = [&](int s2, int buf) {
    const int pos = s2 >> 2;
    const int cc  = (s2 & 3) << 5;
    const int kh  = (pos * 11) >> 5;   // pos/3 for pos in [0,9)
    const int kw  = pos - kh * 3;
    // A: weights, k-contiguous bf16 -> async global->LDS (ASYNCcnt-tracked)
    {
      const unsigned short* g =
          wgtBf + (size_t)(m0 + lr) * KTOT + pos * CIN + cc + lh * 16;
      unsigned long long ga = (unsigned long long)(uintptr_t)g;
      unsigned lds = ldsABase + (unsigned)buf * TILEB;
      asm volatile(
          "global_load_async_to_lds_b128 %0, %1, off\n\t"
          "global_load_async_to_lds_b128 %0, %1, off offset:16"
          :: "v"(lds), "v"(ga) : "memory");
    }
    // B: branch-free constant-stride gather (stride HW elems per channel)
    {
      const unsigned short* s =
          dataBf + (((size_t)(b * CIN + cc + lh * 16) * IH) + (ohc + kh)) * IW +
          (owc + kw);
      unsigned int* dst32 = (unsigned int*)(&sB[buf][0] + lr * LDT + lh * 16);
#pragma unroll
      for (int e = 0; e < 8; e++) {
        unsigned lo = s[(size_t)(2 * e)     * HW];
        unsigned hi = s[(size_t)(2 * e + 1) * HW];
        dst32[e] = lo | (hi << 16);
      }
    }
  };

  stage(0, 0);

  for (int s = 0; s < NSTEP; ++s) {
    const int p = s & 1;
    asm volatile("s_wait_asynccnt 0x0" ::: "memory");
    __syncthreads();   // compiler adds s_wait_dscnt: buffer p fully written

    // fragment loads from buffer p (CDNA5 16-bit 16x32 layout)
    FragU af[2], bfr[4];
#pragma unroll
    for (int tm = 0; tm < 2; tm++) {
      const unsigned short* base =
          &sA[p][0] + (wm * 32 + tm * 16 + lrow) * LDT + half * 8;
      af[tm].q[0] = *(const u32x4*)(base);
      af[tm].q[1] = *(const u32x4*)(base + 16);
    }
#pragma unroll
    for (int tn = 0; tn < 4; tn++) {
      const unsigned short* base =
          &sB[p][0] + (wn * 64 + tn * 16 + lrow) * LDT + half * 8;
      bfr[tn].q[0] = *(const u32x4*)(base);
      bfr[tn].q[1] = *(const u32x4*)(base + 16);
    }

    // kick next tile's global traffic into the other buffer (overlaps WMMA)
    if (s + 1 < NSTEP) stage(s + 1, 1 - p);

#pragma unroll
    for (int tm = 0; tm < 2; tm++)
#pragma unroll
      for (int tn = 0; tn < 4; tn++)
        acc[tm][tn] = __builtin_amdgcn_wmma_f32_16x16x32_bf16(
            false, af[tm].v, false, bfr[tn].v,
            (short)0, acc[tm][tn], false, false);
  }

  // ---- epilogue: C/D layout -> out[(b*256+m)*3844 + n] ----
  const int nColBase = n0 + wn * 64 + lrow;
#pragma unroll
  for (int tm = 0; tm < 2; tm++) {
#pragma unroll
    for (int tn = 0; tn < 4; tn++) {
      const int n = nColBase + tn * 16;
      if (n >= NSP) continue;
      const int mBase = m0 + wm * 32 + tm * 16 + half * 8;
      float* o = out + (size_t)b * COUT * NSP + n;
#pragma unroll
      for (int r = 0; r < 8; r++)
        o[(size_t)(mBase + r) * NSP] = acc[tm][tn][r];
    }
  }
}

extern "C" void kernel_launch(void* const* d_in, const int* in_sizes, int n_in,
                              void* d_out, int out_size, void* d_ws, size_t ws_size,
                              hipStream_t stream) {
  (void)in_sizes; (void)n_in; (void)out_size; (void)ws_size;
  const float* data = (const float*)d_in[0];   // [32,128,64,64] f32
  const float* wgt  = (const float*)d_in[1];   // [256,128,3,3]  f32
  float*       out  = (float*)d_out;           // [32,256,62,62] f32

  unsigned short* wsW = (unsigned short*)d_ws;   // bf16 weights [o][pos][c]
  unsigned short* wsD = wsW + WGT_ELEMS;         // bf16 data NCHW

  wgt_to_bf16<<<WGT_ELEMS / 256, 256, 0, stream>>>(wgt, wsW);
  data_to_bf16<<<DATA_ELEMS / (256 * 4), 256, 0, stream>>>(data, wsD);

  dim3 block(256);
  dim3 grid((NSP + BN - 1) / BN, COUT / BM, 32);
  conv2d_wmma_bf16<<<grid, block, 0, stream>>>(wsD, wsW, out);
}
